// GlobalAttention_9182640079544
// MI455X (gfx1250) — matrix operand order repaired and checked by reading, not compiled, not run on hardware
//
#include <hip/hip_runtime.h>
#include <hip/hip_bf16.h>

// ---------------------------------------------------------------------------
// Problem constants (match reference)
// ---------------------------------------------------------------------------
#define BATCH 4
#define SEQ   2048
#define DMODEL 1024
#define NTILES_PER_BATCH (SEQ / 16)   // 128

typedef __attribute__((ext_vector_type(16))) __bf16 v16bf;
typedef __attribute__((ext_vector_type(8)))  __bf16 v8bf;
typedef __attribute__((ext_vector_type(8)))  float  v8f;

#define NEG_BIG (-3.0e38f)

// Workspace layout (bytes)
#define XB_OFF  (0ull)               // bf16 x        : 16 MB
#define QB_OFF  (16ull << 20)        // bf16 Q        : 16 MB
#define KB_OFF  (32ull << 20)        // bf16 K+pos    : 16 MB
#define VT_OFF  (48ull << 20)        // bf16 V^T      : 16 MB  [B][D][S]
#define WQ_OFF  (64ull << 20)        // bf16 Wq       : 2 MB
#define WK_OFF  (66ull << 20)        // bf16 Wk       : 2 MB
#define WV_OFF  (68ull << 20)        // bf16 Wv       : 2 MB
#define RS_OFF  (70ull << 20)        // f32 rowscale  : 32 KB
#define PR_OFF  (71ull << 20)        // bf16 probs    : 32 MB  [B][S][S]

// A-fragment loader: 16x32 bf16 tile, contiguous K order (matches B loads).
// lane<16 : row = base_r, K [k0+0..15]
// lane>=16: row = base_r, K [k0+16..31]
static __device__ __forceinline__ v16bf load_a(const __bf16* rowp, int k0, int half) {
  const __bf16* ap = rowp + k0 + half * 16;
  v8bf alo = *(const v8bf*)ap;
  v8bf ahi = *(const v8bf*)(ap + 8);
  return __builtin_shufflevector(alo, ahi, 0,1,2,3,4,5,6,7,8,9,10,11,12,13,14,15);
}

// ---------------------------------------------------------------------------
// fp32 -> bf16 conversion
// ---------------------------------------------------------------------------
__global__ __launch_bounds__(256) void cvt_f32_bf16(const float* __restrict__ src,
                                                    __bf16* __restrict__ dst, int n) {
  int i = blockIdx.x * 256 + threadIdx.x;
  if (i < n) dst[i] = (__bf16)src[i];
}

// Zero-fill (16-byte granularity)
__global__ __launch_bounds__(256) void zero_b128(uint4* __restrict__ p, int nvec) {
  int i = blockIdx.x * 256 + threadIdx.x;
  if (i < nvec) p[i] = make_uint4(0u, 0u, 0u, 0u);
}

// ---------------------------------------------------------------------------
// Projection GEMM: y = x @ W^T + b  (+ optional pos-add, optional transpose-out)
// Register-blocked: each wave computes a 32x64 tile (2 M-sub x 4 N-sub, 8 acc).
// Per K=32 step: 2 A-loads + 4 B-loads feed 8 WMMAs.
// mode 0: Q   -> out[row*D+col]
// mode 1: K'  -> out[row*D+col] = k + x + P       (pos folded into K)
// mode 2: V^T -> out[(b*D+col)*S + s]             (transposed store)
// grid: (M/32, D/512), block 256 (8 waves, 64 cols each)
// ---------------------------------------------------------------------------
__global__ __launch_bounds__(256) void proj_gemm(const __bf16* __restrict__ xb,
                                                 const __bf16* __restrict__ wb,
                                                 const float* __restrict__ bias,
                                                 const float* __restrict__ x,
                                                 const float* __restrict__ P,
                                                 __bf16* __restrict__ out,
                                                 int mode) {
  const int lid  = threadIdx.x & 31;
  const int wv   = threadIdx.x >> 5;
  const int half = lid >> 4;
  const int r    = lid & 15;
  const int m0   = blockIdx.x * 32;
  const int n0   = blockIdx.y * 512 + wv * 64;

  v8f acc[2][4];
#pragma unroll
  for (int i = 0; i < 2; ++i)
#pragma unroll
    for (int j = 0; j < 4; ++j) acc[i][j] = (v8f){};

  const __bf16* arow0 = xb + (size_t)(m0 + r) * DMODEL;
  const __bf16* arow1 = xb + (size_t)(m0 + 16 + r) * DMODEL;
  const __bf16* brow[4];
#pragma unroll
  for (int j = 0; j < 4; ++j) brow[j] = wb + (size_t)(n0 + 16 * j + r) * DMODEL;

  for (int k0 = 0; k0 < DMODEL; k0 += 32) {
    v16bf a0 = load_a(arow0, k0, half);
    v16bf a1 = load_a(arow1, k0, half);
#pragma unroll
    for (int j = 0; j < 4; ++j) {
      v16bf b = *(const v16bf*)(brow[j] + k0 + half * 16);
      acc[0][j] = __builtin_amdgcn_wmma_f32_16x16x32_bf16(false, a0, false, b,
                                                          (short)0, acc[0][j], false, false);
      acc[1][j] = __builtin_amdgcn_wmma_f32_16x16x32_bf16(false, a1, false, b,
                                                          (short)0, acc[1][j], false, false);
    }
  }

#pragma unroll
  for (int i = 0; i < 2; ++i) {
#pragma unroll
    for (int j = 0; j < 4; ++j) {
      const int col = n0 + 16 * j + r;
      const float bcol = bias[col];
#pragma unroll
      for (int e = 0; e < 8; ++e) {
        const int row = m0 + 16 * i + e + half * 8;   // global row in [0, B*S)
        float y = acc[i][j][e] + bcol;
        if (mode == 1) {
          const int s = row & (SEQ - 1);
          y += x[(size_t)row * DMODEL + col] + P[(size_t)s * DMODEL + col];
        }
        if (mode == 2) {
          const int bidx = row >> 11;                 // row / SEQ
          const int s    = row & (SEQ - 1);
          out[((size_t)bidx * DMODEL + col) * SEQ + s] = (__bf16)y;
        } else {
          out[(size_t)row * DMODEL + col] = (__bf16)y;
        }
      }
    }
  }
}

// ---------------------------------------------------------------------------
// Scores + causal softmax. One wave per 16-row query tile.
// Pass 1: S = Q K'^T / sqrt(D), 4 key-subtiles per Q-fragment load (64 t-cols),
//         causal mask, store raw bf16, track rowmax.
// Pass 2: reload, exp(s - max), accumulate row sum, store unnormalized probs.
// Row scales (1/sum) applied in the PV epilogue.
// grid: B*S/16/8 blocks of 256 (8 waves)
// ---------------------------------------------------------------------------
__global__ __launch_bounds__(256) void attn_scores(const __bf16* __restrict__ Qb,
                                                   const __bf16* __restrict__ Kkb,
                                                   __bf16* __restrict__ probs,
                                                   float* __restrict__ rowscale) {
  const int lid  = threadIdx.x & 31;
  const int wv   = threadIdx.x >> 5;
  const int half = lid >> 4;
  const int r    = lid & 15;

  const int qt   = blockIdx.x * 8 + wv;          // query tile id in [0, B*S/16)
  const int bidx = qt >> 7;                      // / NTILES_PER_BATCH
  const int i0   = (qt & (NTILES_PER_BATCH - 1)) * 16;
  const int mrow = bidx * SEQ + i0;
  const float scale = 0.03125f;                  // 1/sqrt(1024)

  float m[8];
#pragma unroll
  for (int e = 0; e < 8; ++e) m[e] = NEG_BIG;

  const __bf16* arow = Qb + (size_t)(mrow + r) * DMODEL;

  // ---- pass 1: raw scores + running row max (4 key-subtiles per A load) ----
  for (int t0 = 0; t0 <= i0; t0 += 64) {
    v8f acc[4];
#pragma unroll
    for (int j = 0; j < 4; ++j) acc[j] = (v8f){};

    for (int k0 = 0; k0 < DMODEL; k0 += 32) {
      v16bf a = load_a(arow, k0, half);
#pragma unroll
      for (int j = 0; j < 4; ++j) {
        if (t0 + 16 * j <= i0) {                 // uniform (scalar) guard
          const __bf16* brow = Kkb + (size_t)(bidx * SEQ + t0 + 16 * j + r) * DMODEL;
          v16bf b = *(const v16bf*)(brow + k0 + half * 16);
          acc[j] = __builtin_amdgcn_wmma_f32_16x16x32_bf16(false, a, false, b,
                                                           (short)0, acc[j], false, false);
        }
      }
    }
#pragma unroll
    for (int j = 0; j < 4; ++j) {
      if (t0 + 16 * j <= i0) {
#pragma unroll
        for (int e = 0; e < 8; ++e) {
          const int srow = i0 + e + half * 8;
          const int tcol = t0 + 16 * j + r;
          float sc = acc[j][e] * scale;
          if (tcol > srow) sc = NEG_BIG;         // causal mask (diagonal tile)
          m[e] = fmaxf(m[e], sc);
          probs[(size_t)(bidx * SEQ + srow) * SEQ + tcol] = (__bf16)sc;
        }
      }
    }
  }
  // reduce row max across the 16 lanes of each half-wave
#pragma unroll
  for (int e = 0; e < 8; ++e) {
    for (int off = 1; off < 16; off <<= 1)
      m[e] = fmaxf(m[e], __shfl_xor(m[e], off, 32));
  }

  // ---- pass 2: exp + row sums, write unnormalized probs ----
  float l[8];
#pragma unroll
  for (int e = 0; e < 8; ++e) l[e] = 0.0f;

  for (int t0 = 0; t0 <= i0; t0 += 16) {
#pragma unroll
    for (int e = 0; e < 8; ++e) {
      const int srow = i0 + e + half * 8;
      const int tcol = t0 + r;
      const size_t idx = (size_t)(bidx * SEQ + srow) * SEQ + tcol;
      float p = __expf((float)probs[idx] - m[e]);
      l[e] += p;
      probs[idx] = (__bf16)p;
    }
  }
#pragma unroll
  for (int e = 0; e < 8; ++e) {
    for (int off = 1; off < 16; off <<= 1)
      l[e] += __shfl_xor(l[e], off, 32);
  }
  if (r == 0) {
#pragma unroll
    for (int e = 0; e < 8; ++e)
      rowscale[bidx * SEQ + i0 + e + half * 8] = 1.0f / l[e];
  }
}

// ---------------------------------------------------------------------------
// out = (probs @ V) * rowscale, fp32 output [B][S][D]
// Register-blocked: each wave computes 16x64 (4 d-subtiles per A load).
// A = probs [b][16 rows][t] bf16, B-operand = V^T [b][d][t] (contiguous in t)
// grid: (B*S/16, D/512), block 256 (8 waves, 64 cols each)
// ---------------------------------------------------------------------------
__global__ __launch_bounds__(256) void attn_pv(const __bf16* __restrict__ probs,
                                               const __bf16* __restrict__ Vtb,
                                               const float* __restrict__ rowscale,
                                               float* __restrict__ out) {
  const int lid  = threadIdx.x & 31;
  const int wv   = threadIdx.x >> 5;
  const int half = lid >> 4;
  const int r    = lid & 15;

  const int qt   = blockIdx.x;
  const int bidx = qt >> 7;
  const int i0   = (qt & (NTILES_PER_BATCH - 1)) * 16;
  const int n0   = blockIdx.y * 512 + wv * 64;

  const __bf16* arow = probs + (size_t)(bidx * SEQ + i0 + r) * SEQ;
  const __bf16* brow[4];
#pragma unroll
  for (int j = 0; j < 4; ++j)
    brow[j] = Vtb + ((size_t)bidx * DMODEL + n0 + 16 * j + r) * SEQ;

  v8f acc[4];
#pragma unroll
  for (int j = 0; j < 4; ++j) acc[j] = (v8f){};

  const int tend = i0 + 16;                      // causal extent (exclusive)
  for (int t0 = 0; t0 < tend; t0 += 32) {        // tail chunk reads zeros
    v16bf a = load_a(arow, t0, half);
#pragma unroll
    for (int j = 0; j < 4; ++j) {
      v16bf b = *(const v16bf*)(brow[j] + t0 + half * 16);
      acc[j] = __builtin_amdgcn_wmma_f32_16x16x32_bf16(false, a, false, b,
                                                       (short)0, acc[j], false, false);
    }
  }

#pragma unroll
  for (int j = 0; j < 4; ++j) {
    const int d = n0 + 16 * j + r;
#pragma unroll
    for (int e = 0; e < 8; ++e) {
      const int s = i0 + e + half * 8;
      out[(size_t)(bidx * SEQ + s) * DMODEL + d] = acc[j][e] * rowscale[bidx * SEQ + s];
    }
  }
}

// ---------------------------------------------------------------------------
// Launch
// ---------------------------------------------------------------------------
extern "C" void kernel_launch(void* const* d_in, const int* in_sizes, int n_in,
                              void* d_out, int out_size, void* d_ws, size_t ws_size,
                              hipStream_t stream) {
  const float* x  = (const float*)d_in[0];
  const float* Wq = (const float*)d_in[1];
  const float* bq = (const float*)d_in[2];
  const float* Wk = (const float*)d_in[3];
  const float* bk = (const float*)d_in[4];
  const float* Wv = (const float*)d_in[5];
  const float* bv = (const float*)d_in[6];
  const float* P  = (const float*)d_in[7];
  float* out = (float*)d_out;

  char* ws = (char*)d_ws;
  __bf16* xb   = (__bf16*)(ws + XB_OFF);
  __bf16* Qb   = (__bf16*)(ws + QB_OFF);
  __bf16* Kkb  = (__bf16*)(ws + KB_OFF);
  __bf16* Vtb  = (__bf16*)(ws + VT_OFF);
  __bf16* Wqb  = (__bf16*)(ws + WQ_OFF);
  __bf16* Wkb  = (__bf16*)(ws + WK_OFF);
  __bf16* Wvb  = (__bf16*)(ws + WV_OFF);
  float*  rsc  = (float*)(ws + RS_OFF);
  __bf16* prob = (__bf16*)(ws + PR_OFF);

  const int nX = BATCH * SEQ * DMODEL;     // 8,388,608
  const int nW = DMODEL * DMODEL;          // 1,048,576

  // 1) precision conversion (bf16 WMMA path with fp32 accumulation)
  cvt_f32_bf16<<<(nX + 255) / 256, 256, 0, stream>>>(x,  xb,  nX);
  cvt_f32_bf16<<<(nW + 255) / 256, 256, 0, stream>>>(Wq, Wqb, nW);
  cvt_f32_bf16<<<(nW + 255) / 256, 256, 0, stream>>>(Wk, Wkb, nW);
  cvt_f32_bf16<<<(nW + 255) / 256, 256, 0, stream>>>(Wv, Wvb, nW);

  // 2) zero probs buffer (so causal/tail regions read as 0 in the PV GEMM)
  const int prVecs = (BATCH * SEQ * SEQ * 2) / 16;   // bf16 bytes / 16
  zero_b128<<<(prVecs + 255) / 256, 256, 0, stream>>>((uint4*)prob, prVecs);

  // 3) fused QKV projections (pos folded into K'; V stored transposed)
  dim3 pgrid(BATCH * SEQ / 32, DMODEL / 512);
  proj_gemm<<<pgrid, 256, 0, stream>>>(xb, Wqb, bq, x, P, Qb,  0);
  proj_gemm<<<pgrid, 256, 0, stream>>>(xb, Wkb, bk, x, P, Kkb, 1);
  proj_gemm<<<pgrid, 256, 0, stream>>>(xb, Wvb, bv, x, P, Vtb, 2);

  // 4) scores + causal softmax (probs materialized in L2-resident scratch)
  attn_scores<<<(BATCH * SEQ / 16) / 8, 256, 0, stream>>>(Qb, Kkb, prob, rsc);

  // 5) out = softmax(scores) @ V
  dim3 vgrid(BATCH * SEQ / 16, DMODEL / 512);
  attn_pv<<<vgrid, 256, 0, stream>>>(prob, Vtb, rsc, out);
}
